// DimeNetDiabat_27805618274370
// MI455X (gfx1250) — matrix operand order repaired and checked by reading, not compiled, not run on hardware
//
#include <hip/hip_runtime.h>
#include <hip/hip_bf16.h>

typedef __attribute__((ext_vector_type(16))) _Float16 v16h;
typedef __attribute__((ext_vector_type(8)))  float    v8f;

#define PI_F 3.14159265358979323846f

__device__ __forceinline__ float swishf(float v) { return v / (1.0f + __expf(-v)); }

// DimeNet envelope, p=6: 1/x - 28 x^5 + 48 x^6 - 21 x^7, zero for x>=1
__device__ __forceinline__ float envf(float x) {
    if (x >= 1.0f) return 0.0f;
    float x2 = x * x, x4 = x2 * x2, x5 = x4 * x;
    return 1.0f / x + (-28.0f) * x5 + 48.0f * x5 * x + (-21.0f) * x5 * x2;
}

// ---------------------------------------------------------------------------
// WMMA GEMM: out[n_rows x M] = epilogue(A[n_rows x lda (gathered?)] @ W[K x M])
//   MT = M/16 (4 or 8). K multiple of 32, K <= 128.
//   rix      : optional row gather (A row = rix[row])
//   acc_prev : add existing out (multi-pass K accumulation)
//   bias     : optional [M]
//   do_swish : apply swish after (acc + bias + prev)
//   resid    : optional [n_rows x M], added after activation (out==resid OK)
// 256 threads = 8 waves; wave strip = 32 rows (two 16-row A fragments sharing
// every B fragment -> 2 WMMA per ds_load_b128 pair); block covers 256 rows.
// ---------------------------------------------------------------------------
template <int MT>
__global__ __launch_bounds__(256) void gemm_wmma_f16(
    const float* __restrict__ A, const int* __restrict__ rix, int lda,
    const float* __restrict__ W, int K,
    const float* __restrict__ bias, const float* __restrict__ resid,
    float* __restrict__ out, int n_rows, int acc_prev, int do_swish)
{
    constexpr int M    = MT * 16;
    constexpr int LSTR = 144;  // padded K stride (halfs); rows 288B, 32B aligned
    __shared__ __align__(32) _Float16 sWt[M * LSTR];

    const int tid = threadIdx.x;
    // Stage W (K x M row-major) transposed into LDS as f16: sWt[n][k].
    // 4 K-consecutive halfs per thread -> v_cvt_pk_f16_f32 x2 + ds_store_b64.
    for (int idx = tid; idx < M * (K >> 2); idx += 256) {
        const int n  = idx % M;
        const int kc = (idx / M) << 2;
        float w0 = W[(size_t)(kc + 0) * M + n];
        float w1 = W[(size_t)(kc + 1) * M + n];
        float w2 = W[(size_t)(kc + 2) * M + n];
        float w3 = W[(size_t)(kc + 3) * M + n];
        union { _Float16 h[4]; uint2 u; } p;
        p.h[0] = (_Float16)w0; p.h[1] = (_Float16)w1;
        p.h[2] = (_Float16)w2; p.h[3] = (_Float16)w3;
        *(uint2*)(&sWt[n * LSTR + kc]) = p.u;
    }
    __syncthreads();

    const int wave = tid >> 5, lane = tid & 31;
    const int half_sel = lane >> 4, nn = lane & 15;
    const int rowBase = blockIdx.x * 256 + wave * 32;
    if (rowBase >= n_rows) return;  // wave-uniform: EXEC stays all-ones for WMMA

    const int r0 = rowBase + nn;
    int r1 = rowBase + 16 + nn;
    if (r1 >= n_rows) r1 = n_rows - 1;  // clamp (rows are multiples of 32 here)
    const float* a0 = A + (size_t)(rix ? rix[r0] : r0) * lda + half_sel * 16;
    const float* a1 = A + (size_t)(rix ? rix[r1] : r1) * lda + half_sel * 16;

    v8f acc0[MT], acc1[MT];
#pragma unroll
    for (int t = 0; t < MT; ++t)
#pragma unroll
        for (int q = 0; q < 8; ++q) { acc0[t][q] = 0.0f; acc1[t][q] = 0.0f; }

    // A fragment: contiguous k order, same as the B fragment. Element j -> K = half*16 + j
    float4 q00, q01, q02, q03, q10, q11, q12, q13;
    auto LD = [](const float* p, int k0, float4& x0, float4& x1, float4& x2, float4& x3) {
        x0 = *(const float4*)(p + k0);
        x1 = *(const float4*)(p + k0 + 4);
        x2 = *(const float4*)(p + k0 + 8);
        x3 = *(const float4*)(p + k0 + 12);
    };
    auto CVT = [](float4 x0, float4 x1, float4 x2, float4 x3) -> v16h {
        v16h v;
        v[0]  = (_Float16)x0.x; v[1]  = (_Float16)x0.y; v[2]  = (_Float16)x0.z; v[3]  = (_Float16)x0.w;
        v[4]  = (_Float16)x1.x; v[5]  = (_Float16)x1.y; v[6]  = (_Float16)x1.z; v[7]  = (_Float16)x1.w;
        v[8]  = (_Float16)x2.x; v[9]  = (_Float16)x2.y; v[10] = (_Float16)x2.z; v[11] = (_Float16)x2.w;
        v[12] = (_Float16)x3.x; v[13] = (_Float16)x3.y; v[14] = (_Float16)x3.z; v[15] = (_Float16)x3.w;
        return v;
    };

    LD(a0, 0, q00, q01, q02, q03);
    LD(a1, 0, q10, q11, q12, q13);
    v16h va0 = CVT(q00, q01, q02, q03);
    v16h va1 = CVT(q10, q11, q12, q13);

    for (int k0 = 0; k0 < K; k0 += 32) {
        const bool more = (k0 + 32) < K;
        if (more) {  // issue next-step A loads; they complete under the WMMA chain
            LD(a0, k0 + 32, q00, q01, q02, q03);
            LD(a1, k0 + 32, q10, q11, q12, q13);
        }
        // B fragment: element j -> K = half*16 + j, N = t*16+nn. 32B contiguous in LDS.
        v16h vb = *(const v16h*)(&sWt[nn * LSTR + k0 + half_sel * 16]);
#pragma unroll
        for (int t = 0; t < MT; ++t) {
            v16h vbn;
            if (t + 1 < MT)
                vbn = *(const v16h*)(&sWt[((t + 1) * 16 + nn) * LSTR + k0 + half_sel * 16]);
            acc0[t] = __builtin_amdgcn_wmma_f32_16x16x32_f16(
                false, va0, false, vb, (short)0, acc0[t], false, false);
            acc1[t] = __builtin_amdgcn_wmma_f32_16x16x32_f16(
                false, va1, false, vb, (short)0, acc1[t], false, false);
            if (t + 1 < MT) vb = vbn;
        }
        if (more) {  // convert after the WMMA chain (wait_loadcnt lands here)
            va0 = CVT(q00, q01, q02, q03);
            va1 = CVT(q10, q11, q12, q13);
        }
    }

#pragma unroll
    for (int t = 0; t < MT; ++t) {
        const int col = t * 16 + nn;
        const float b = bias ? bias[col] : 0.0f;
#pragma unroll
        for (int r = 0; r < 8; ++r) {
            {   // strip 0: C/D layout M = r + 8*half
                const int row = rowBase + half_sel * 8 + r;
                const size_t off = (size_t)row * M + col;
                float v = acc0[t][r] + b;
                if (acc_prev) v += out[off];
                if (do_swish) v = swishf(v);
                if (resid)    v += resid[off];
                out[off] = v;
            }
            {   // strip 1
                const int row = rowBase + 16 + half_sel * 8 + r;
                if (row < n_rows) {
                    const size_t off = (size_t)row * M + col;
                    float v = acc1[t][r] + b;
                    if (acc_prev) v += out[off];
                    if (do_swish) v = swishf(v);
                    if (resid)    v += resid[off];
                    out[off] = v;
                }
            }
        }
    }
}

// ---------------------------------------------------------------------------
// Elementwise / glue kernels
// ---------------------------------------------------------------------------
__global__ void k_edge_geom(const float* __restrict__ xyz, const int* __restrict__ nbr,
                            float* __restrict__ xs, float* __restrict__ e_rbf, int ne) {
    int e = blockIdx.x * blockDim.x + threadIdx.x;
    if (e >= ne) return;
    int i = nbr[2 * e], j = nbr[2 * e + 1];
    float dx = xyz[3 * i] - xyz[3 * j];
    float dy = xyz[3 * i + 1] - xyz[3 * j + 1];
    float dz = xyz[3 * i + 2] - xyz[3 * j + 2];
    float d = sqrtf(dx * dx + dy * dy + dz * dz);
    float x = d * (1.0f / 5.0f);
    float env = envf(x);
    xs[e] = x;
#pragma unroll
    for (int r = 0; r < 6; ++r)
        e_rbf[(size_t)e * 6 + r] = env * __sinf(PI_F * (float)(r + 1) * x);
}

__global__ void k_angle_geom(const float* __restrict__ xyz, const int* __restrict__ ang,
                             const int* __restrict__ kj, const float* __restrict__ xs,
                             float* __restrict__ a_sbf, int nw) {
    int w = blockIdx.x * blockDim.x + threadIdx.x;
    if (w >= nw) return;
    int i = ang[3 * w], j = ang[3 * w + 1], k = ang[3 * w + 2];
    float ax = xyz[3 * i] - xyz[3 * j], ay = xyz[3 * i + 1] - xyz[3 * j + 1], az = xyz[3 * i + 2] - xyz[3 * j + 2];
    float bx = xyz[3 * k] - xyz[3 * j], by = xyz[3 * k + 1] - xyz[3 * j + 1], bz = xyz[3 * k + 2] - xyz[3 * j + 2];
    float ct = ax * bx + ay * by + az * bz;
    float cx = ay * bz - az * by, cy = az * bx - ax * bz, cz = ax * by - ay * bx;
    float st = sqrtf(cx * cx + cy * cy + cz * cz + 1e-12f);
    float alpha = atan2f(st, ct);
    float xk = xs[kj[w]];
    float env = envf(xk);
    float rad[6];
#pragma unroll
    for (int s = 0; s < 6; ++s) rad[s] = env * __sinf(PI_F * (float)(s + 1) * xk);
    float* o = a_sbf + (size_t)w * 64;
#pragma unroll
    for (int l = 0; l < 7; ++l) {
        float c = __cosf((float)l * alpha);
#pragma unroll
        for (int s = 0; s < 6; ++s) o[l * 6 + s] = c * rad[s];
    }
#pragma unroll
    for (int q = 42; q < 64; ++q) o[q] = 0.0f;  // zero-pad K to 64 for WMMA
}

// rbf_e = swish(e_rbf @ emb_rbf_w + b)   [NE x 128], K=6 scalar
__global__ void k_rbf_emb(const float* __restrict__ e_rbf, const float* __restrict__ w6,
                          const float* __restrict__ b, float* __restrict__ out, int ne) {
    int idx = blockIdx.x * blockDim.x + threadIdx.x;
    int e = idx >> 7, n = idx & 127;
    if (e >= ne) return;
    float s = b[n];
#pragma unroll
    for (int r = 0; r < 6; ++r) s += e_rbf[(size_t)e * 6 + r] * w6[r * 128 + n];
    out[(size_t)idx] = swishf(s);
}

__global__ void k_rix(const int* __restrict__ z, const int* __restrict__ nbr,
                      int* __restrict__ rj, int* __restrict__ ri, int ne) {
    int e = blockIdx.x * blockDim.x + threadIdx.x;
    if (e >= ne) return;
    rj[e] = z[nbr[2 * e + 1]];
    ri[e] = z[nbr[2 * e]];
}

// Wr[6][128] = iw_rbf1[6][8] @ iw_rbf2[8][128]
__global__ void k_fold_rbf(const float* __restrict__ w1, const float* __restrict__ w2,
                           float* __restrict__ wr) {
    int idx = blockIdx.x * blockDim.x + threadIdx.x;
    if (idx >= 6 * 128) return;
    int r = idx / 128, n = idx - r * 128;
    float s = 0.0f;
#pragma unroll
    for (int b = 0; b < 8; ++b) s += w1[r * 8 + b] * w2[b * 128 + n];
    wr[idx] = s;
}

// Wsbf[64][64]: rows 0..41 = iw_sbf1[42][8] @ iw_sbf2[8][64]; rows 42..63 = 0
__global__ void k_fold_sbf(const float* __restrict__ w1, const float* __restrict__ w2,
                           float* __restrict__ ws) {
    int idx = blockIdx.x * blockDim.x + threadIdx.x;
    if (idx >= 64 * 64) return;
    int s = idx >> 6, n = idx & 63;
    float v = 0.0f;
    if (s < 42) {
#pragma unroll
        for (int b = 0; b < 8; ++b) v += w1[s * 8 + b] * w2[b * 64 + n];
    }
    ws[idx] = v;
}

// x_kj[e,n] *= (e_rbf[e] . Wr[:,n])
__global__ void k_apply_rbf(const float* __restrict__ e_rbf, const float* __restrict__ wr,
                            float* __restrict__ x_kj, int ne) {
    int idx = blockIdx.x * blockDim.x + threadIdx.x;
    int e = idx >> 7, n = idx & 127;
    if (e >= ne) return;
    float s = 0.0f;
#pragma unroll
    for (int r = 0; r < 6; ++r) s += e_rbf[(size_t)e * 6 + r] * wr[r * 128 + n];
    x_kj[(size_t)idx] *= s;
}

// agg[ji_idx[w]] += x_kj_e[kj_idx[w]] * sbf_p[w]   (64 wide)
__global__ void k_triplet(const float* __restrict__ xkje, const float* __restrict__ sbfp,
                          const int* __restrict__ kj, const int* __restrict__ ji,
                          float* __restrict__ agg, int nw) {
    int idx = blockIdx.x * blockDim.x + threadIdx.x;
    int w = idx >> 6, n = idx & 63;
    if (w >= nw) return;
    float v = xkje[(size_t)kj[w] * 64 + n] * sbfp[(size_t)idx];
    atomicAdd(agg + (size_t)ji[w] * 64 + n, v);
}

// atoms[nbr0[e]] += (e_rbf[e] . ow_rbf[:,n]) * m[e,n]
__global__ void k_out_scatter(const float* __restrict__ e_rbf, const float* __restrict__ wr,
                              const float* __restrict__ mbuf, const int* __restrict__ nbr,
                              float* __restrict__ atoms, int ne) {
    int idx = blockIdx.x * blockDim.x + threadIdx.x;
    int e = idx >> 7, n = idx & 127;
    if (e >= ne) return;
    float s = 0.0f;
#pragma unroll
    for (int r = 0; r < 6; ++r) s += e_rbf[(size_t)e * 6 + r] * wr[r * 128 + n];
    atomicAdd(atoms + (size_t)nbr[2 * e] * 128 + n, s * mbuf[(size_t)idx]);
}

// atomwise[a] += atoms[a,:] . wout
__global__ void k_out_dot(const float* __restrict__ atoms, const float* __restrict__ wout,
                          float* __restrict__ atomwise, int na) {
    int a = blockIdx.x * blockDim.x + threadIdx.x;
    if (a >= na) return;
    float s = 0.0f;
    for (int n = 0; n < 128; ++n) s += atoms[(size_t)a * 128 + n] * wout[n];
    atomwise[a] += s;
}

__global__ void k_seg(const float* __restrict__ atomwise, const int* __restrict__ mol,
                      float* __restrict__ E, int na, int ng) {
    int a = blockIdx.x * blockDim.x + threadIdx.x;
    if (a >= na) return;
    int g = mol[a];
#pragma unroll
    for (int k = 0; k < 3; ++k) atomicAdd(&E[k * ng + g], atomwise[(size_t)k * na + a]);
}

// 2x2 symmetric eig, ascending (matches eigvalsh)
__global__ void k_eig(const float* __restrict__ E, float* __restrict__ out, int ng) {
    int g = blockIdx.x * blockDim.x + threadIdx.x;
    if (g >= ng) return;
    float d0 = E[g], d1 = E[ng + g], lam = E[2 * ng + g];
    float mean = 0.5f * (d0 + d1), dd = 0.5f * (d0 - d1);
    float s = sqrtf(dd * dd + lam * lam);
    out[2 * g] = mean - s;
    out[2 * g + 1] = mean + s;
}

// ---------------------------------------------------------------------------
extern "C" void kernel_launch(void* const* d_in, const int* in_sizes, int n_in,
                              void* d_out, int out_size, void* d_ws, size_t ws_size,
                              hipStream_t stream) {
    const float* xyz       = (const float*)d_in[0];
    const float* emb_z     = (const float*)d_in[1];
    const float* emb_rbf_w = (const float*)d_in[2];
    const float* emb_rbf_b = (const float*)d_in[3];
    const float* emb_w     = (const float*)d_in[4];
    const float* emb_b     = (const float*)d_in[5];
    const float* iw_ji     = (const float*)d_in[6];
    const float* ib_ji     = (const float*)d_in[7];
    const float* iw_kj     = (const float*)d_in[8];
    const float* ib_kj     = (const float*)d_in[9];
    const float* iw_rbf1   = (const float*)d_in[10];
    const float* iw_rbf2   = (const float*)d_in[11];
    const float* iw_sbf1   = (const float*)d_in[12];
    const float* iw_sbf2   = (const float*)d_in[13];
    const float* iw_down   = (const float*)d_in[14];
    const float* iw_up     = (const float*)d_in[15];
    const float* iw_final  = (const float*)d_in[16];
    const float* ib_final  = (const float*)d_in[17];
    const float* iw_res    = (const float*)d_in[18];
    const float* ib_res    = (const float*)d_in[19];
    const float* ow_rbf    = (const float*)d_in[20];
    const float* ow_dense  = (const float*)d_in[21];
    const float* ob_dense  = (const float*)d_in[22];
    const float* ow_out    = (const float*)d_in[23];
    const int*   z         = (const int*)d_in[24];
    const int*   nbr       = (const int*)d_in[25];
    const int*   ang       = (const int*)d_in[26];
    const int*   kj_idx    = (const int*)d_in[27];
    const int*   ji_idx    = (const int*)d_in[28];
    const int*   mol_idx   = (const int*)d_in[29];

    const int NA = in_sizes[24];
    const int NE = in_sizes[25] / 2;
    const int NW = in_sizes[26] / 3;
    const int NG = out_size / 2;

    // ---- workspace layout ----
    char* wp = (char*)d_ws;
    auto allocf = [&](size_t n) -> float* {
        float* p = (float*)wp; wp += ((n * 4 + 255) / 256) * 256; return p; };
    auto alloci = [&](size_t n) -> int* {
        int* p = (int*)wp; wp += ((n * 4 + 255) / 256) * 256; return p; };

    float* xs      = allocf(NE);
    float* e_rbf   = allocf((size_t)NE * 6);
    float* rbf_e   = allocf((size_t)NE * 128);
    float* mbuf    = allocf((size_t)NE * 128);
    float* x_ji    = allocf((size_t)NE * 128);
    float* x_kj    = allocf((size_t)NE * 128);
    float* hh      = allocf((size_t)NE * 128);
    float* tmp_e   = allocf((size_t)NE * 128);
    float* x_kj_e  = allocf((size_t)NE * 64);
    float* agg     = allocf((size_t)NE * 64);
    float* a_sbf   = allocf((size_t)NW * 64);
    float* sbf_p   = allocf((size_t)NW * 64);
    float* atoms_h = allocf((size_t)NA * 128);
    float* atoms_0 = allocf((size_t)NA * 128);
    float* atoms_1 = allocf((size_t)NA * 128);
    float* atomwise= allocf((size_t)3 * NA);
    float* Esum    = allocf((size_t)3 * NG);
    float* Wr      = allocf(6 * 128);
    float* Wsbf    = allocf(64 * 64);
    int*   rix_j   = alloci(NE);
    int*   rix_i   = alloci(NE);

    auto cdiv = [](long long a, long long b) { return (int)((a + b - 1) / b); };
    auto gemm128 = [&](const float* A, const int* rix, int lda, const float* W, int K,
                       const float* bias, const float* resid, float* out, int n, int accp, int sw) {
        gemm_wmma_f16<8><<<cdiv(n, 256), 256, 0, stream>>>(A, rix, lda, W, K, bias, resid, out, n, accp, sw);
    };
    auto gemm64 = [&](const float* A, const int* rix, int lda, const float* W, int K,
                      const float* bias, const float* resid, float* out, int n, int accp, int sw) {
        gemm_wmma_f16<4><<<cdiv(n, 256), 256, 0, stream>>>(A, rix, lda, W, K, bias, resid, out, n, accp, sw);
    };

    // ---- geometry + bases ----
    k_edge_geom<<<cdiv(NE, 256), 256, 0, stream>>>(xyz, nbr, xs, e_rbf, NE);
    k_angle_geom<<<cdiv(NW, 256), 256, 0, stream>>>(xyz, ang, kj_idx, xs, a_sbf, NW);
    k_rbf_emb<<<cdiv((long long)NE * 128, 256), 256, 0, stream>>>(e_rbf, emb_rbf_w, emb_rbf_b, rbf_e, NE);
    k_rix<<<cdiv(NE, 256), 256, 0, stream>>>(z, nbr, rix_j, rix_i, NE);

    // ---- embedding: m = swish([h_j | h_i | rbf_e] @ emb_w + emb_b), 3 K=128 passes ----
    gemm128(emb_z, rix_j, 128, emb_w,               128, nullptr, nullptr, mbuf, NE, 0, 0);
    gemm128(emb_z, rix_i, 128, emb_w + 128 * 128,   128, nullptr, nullptr, mbuf, NE, 1, 0);
    gemm128(rbf_e, nullptr, 128, emb_w + 256 * 128, 128, emb_b,   nullptr, mbuf, NE, 1, 1);

    hipMemsetAsync(atomwise, 0, (size_t)3 * NA * 4, stream);

    auto out_block = [&](int k, int o) {
        const int ko = k * 5 + o;
        hipMemsetAsync(atoms_h, 0, (size_t)NA * 128 * 4, stream);
        k_out_scatter<<<cdiv((long long)NE * 128, 256), 256, 0, stream>>>(
            e_rbf, ow_rbf + (size_t)ko * 6 * 128, mbuf, nbr, atoms_h, NE);
        const float* base = ow_dense + (size_t)ko * 3 * 128 * 128;
        const float* bb   = ob_dense + (size_t)ko * 3 * 128;
        gemm128(atoms_h, nullptr, 128, base,                 128, bb,       nullptr, atoms_0, NA, 0, 1);
        gemm128(atoms_0, nullptr, 128, base + 128 * 128,     128, bb + 128, nullptr, atoms_1, NA, 0, 1);
        gemm128(atoms_1, nullptr, 128, base + 2 * 128 * 128, 128, bb + 256, nullptr, atoms_0, NA, 0, 1);
        k_out_dot<<<cdiv(NA, 256), 256, 0, stream>>>(atoms_0, ow_out + (size_t)ko * 128,
                                                     atomwise + (size_t)k * NA, NA);
    };

    for (int k = 0; k < 3; ++k) out_block(k, 0);

    // ---- interaction blocks ----
    for (int c = 0; c < 4; ++c) {
        const size_t W128 = (size_t)128 * 128;
        gemm128(mbuf, nullptr, 128, iw_ji + c * W128, 128, ib_ji + c * 128, nullptr, x_ji, NE, 0, 1);
        gemm128(mbuf, nullptr, 128, iw_kj + c * W128, 128, ib_kj + c * 128, nullptr, x_kj, NE, 0, 1);

        k_fold_rbf<<<cdiv(6 * 128, 256), 256, 0, stream>>>(iw_rbf1 + c * 48, iw_rbf2 + c * 8 * 128, Wr);
        k_apply_rbf<<<cdiv((long long)NE * 128, 256), 256, 0, stream>>>(e_rbf, Wr, x_kj, NE);
        gemm64(x_kj, nullptr, 128, iw_down + (size_t)c * 128 * 64, 128, nullptr, nullptr, x_kj_e, NE, 0, 1);

        k_fold_sbf<<<cdiv(64 * 64, 256), 256, 0, stream>>>(iw_sbf1 + c * 42 * 8, iw_sbf2 + c * 8 * 64, Wsbf);
        gemm64(a_sbf, nullptr, 64, Wsbf, 64, nullptr, nullptr, sbf_p, NW, 0, 0);

        hipMemsetAsync(agg, 0, (size_t)NE * 64 * 4, stream);
        k_triplet<<<cdiv((long long)NW * 64, 256), 256, 0, stream>>>(x_kj_e, sbf_p, kj_idx, ji_idx, agg, NW);

        // hh = x_ji + swish(agg @ iw_up)
        gemm128(agg, nullptr, 64, iw_up + (size_t)c * 64 * 128, 64, nullptr, x_ji, hh, NE, 0, 1);

        // hh = res(hh, 0)
        const float* wr00 = iw_res + ((size_t)(c * 3 + 0) * 2 + 0) * W128;
        const float* wr01 = iw_res + ((size_t)(c * 3 + 0) * 2 + 1) * W128;
        const float* br00 = ib_res + ((size_t)(c * 3 + 0) * 2 + 0) * 128;
        const float* br01 = ib_res + ((size_t)(c * 3 + 0) * 2 + 1) * 128;
        gemm128(hh,    nullptr, 128, wr00, 128, br00, nullptr, tmp_e, NE, 0, 1);
        gemm128(tmp_e, nullptr, 128, wr01, 128, br01, hh,      hh,    NE, 0, 1);

        // m = swish(hh @ iw_final + ib_final) + m
        gemm128(hh, nullptr, 128, iw_final + c * W128, 128, ib_final + c * 128, mbuf, mbuf, NE, 0, 1);

        // m = res(res(m,1),2)
        for (int r = 1; r < 3; ++r) {
            const float* w0 = iw_res + ((size_t)(c * 3 + r) * 2 + 0) * W128;
            const float* w1 = iw_res + ((size_t)(c * 3 + r) * 2 + 1) * W128;
            const float* b0 = ib_res + ((size_t)(c * 3 + r) * 2 + 0) * 128;
            const float* b1 = ib_res + ((size_t)(c * 3 + r) * 2 + 1) * 128;
            gemm128(mbuf,  nullptr, 128, w0, 128, b0, nullptr, tmp_e, NE, 0, 1);
            gemm128(tmp_e, nullptr, 128, w1, 128, b1, mbuf,    mbuf,  NE, 0, 1);
        }

        for (int k = 0; k < 3; ++k) out_block(k, c + 1);
    }

    // ---- per-molecule energies + 2x2 eig ----
    hipMemsetAsync(Esum, 0, (size_t)3 * NG * 4, stream);
    k_seg<<<cdiv(NA, 256), 256, 0, stream>>>(atomwise, mol_idx, Esum, NA, NG);
    k_eig<<<cdiv(NG, 256), 256, 0, stream>>>(Esum, (float*)d_out, NG);
}
